// GraphFcNet_26697516712490
// MI455X (gfx1250) — matrix operand order repaired and checked by reading, not compiled, not run on hardware
//
#include <hip/hip_runtime.h>

typedef __attribute__((ext_vector_type(16))) __bf16 v16bf;
typedef __attribute__((ext_vector_type(8)))  float  v8f;

#define N_ROWS 131072
#define BN_EPS 1e-5f
#define COS_EPS 1e-8f

// ---------- helpers ----------
__device__ __forceinline__ unsigned int pk2(float a, float b) {
    unsigned int ua = __builtin_bit_cast(unsigned int, a);
    unsigned int ub = __builtin_bit_cast(unsigned int, b);
    return (ua >> 16) | (ub & 0xffff0000u);   // truncate-to-bf16, packed pair
}
__device__ __forceinline__ unsigned short f2bf(float a) {
    return (unsigned short)(__builtin_bit_cast(unsigned int, a) >> 16);
}
__device__ __forceinline__ float bflo(unsigned int u) {
    return __builtin_bit_cast(float, u << 16);
}
__device__ __forceinline__ float bfhi(unsigned int u) {
    return __builtin_bit_cast(float, u & 0xffff0000u);
}
__device__ __forceinline__ v16bf ldFrag(const uint4* p) {
    union { uint4 q[2]; v16bf b; } u;
    u.q[0] = p[0]; u.q[1] = p[1];
    return u.b;
}
__device__ __forceinline__ v8f wmma_bf(v16bf a, v16bf b, v8f c) {
    return __builtin_amdgcn_wmma_f32_16x16x32_bf16(false, a, false, b, (short)0, c, false, false);
}

// ---------- K0a: W_vf = W_v @ W_f[256:512,:]  (512x256) ----------
__global__ __launch_bounds__(256) void k_wvf(const float* __restrict__ Wv,
                                             const float* __restrict__ Wf,
                                             float* __restrict__ Wvf) {
    int k = blockIdx.x;        // 0..511
    int n = threadIdx.x;       // 0..255
    const float* wvr = Wv + (size_t)k * 256;
    float acc = 0.f;
    for (int j = 0; j < 256; ++j)
        acc += wvr[j] * Wf[(size_t)(256 + j) * 256 + n];   // coalesced over n
    Wvf[(size_t)k * 256 + n] = acc;
}

// ---------- K0c: c = b_f + b_v @ W_f[256:512,:] ----------
__global__ __launch_bounds__(256) void k_cvec(const float* __restrict__ bv,
                                              const float* __restrict__ Wf,
                                              const float* __restrict__ bf,
                                              float* __restrict__ cvec) {
    int n = threadIdx.x;
    float acc = bf[n];
    for (int j = 0; j < 256; ++j)
        acc += bv[j] * Wf[(size_t)(256 + j) * 256 + n];
    cvec[n] = acc;
}

// ---------- K0b: pack B fragments (32x16 bf16, WMMA layout) ----------
// Per ktile fragment: lane l -> column n = l&15; lanes<16 hold K=0..7,16..23,
// lanes>=16 hold K=8..15,24..31; u32 p holds K pair (k0,k0+1).
__global__ __launch_bounds__(256) void k_pack(const float* __restrict__ Wa,
                                              const float* __restrict__ Wf,
                                              const float* __restrict__ Wvf,
                                              unsigned int* __restrict__ waF,
                                              unsigned int* __restrict__ wcF) {
    int gid = blockIdx.x * 256 + threadIdx.x;   // 0..131071
    if (gid < 32768) {                           // W_a: 16 nt x 8 kt
        int p = gid & 7, l = (gid >> 3) & 31, kt = (gid >> 8) & 7, nt = gid >> 11;
        int n  = nt * 16 + (l & 15);
        int k0 = kt * 32 + ((l < 16) ? 0 : 8) + 2 * p + ((p >= 4) ? 8 : 0);
        waF[gid] = pk2(Wa[(size_t)k0 * 256 + n], Wa[(size_t)(k0 + 1) * 256 + n]);
    } else {                                     // W_cat: 16 nt x 24 kt (K=768)
        int g = gid - 32768;
        int p = g & 7, l = (g >> 3) & 31;
        int kt = (g >> 8) % 24, nt = (g >> 8) / 24;
        int n  = nt * 16 + (l & 15);
        int k0 = kt * 32 + ((l < 16) ? 0 : 8) + 2 * p + ((p >= 4) ? 8 : 0);
        float w0 = (k0 < 256) ? Wf[(size_t)k0 * 256 + n] : Wvf[(size_t)(k0 - 256) * 256 + n];
        float w1 = (k0 + 1 < 256) ? Wf[(size_t)(k0 + 1) * 256 + n]
                                  : Wvf[(size_t)(k0 + 1 - 256) * 256 + n];
        wcF[g] = pk2(w0, w1);
    }
}

// ---------- K1: main fused kernel ----------
// 2048 blocks x 256 threads (8 waves, 2 waves/SIMD -> ~512 VGPR budget).
// Block pair (2g, 2g+1) shares the same 128-row group; half = blockIdx&1
// selects channels 0-127 / 128-255. Wave w owns N-tile (half*8 + w) and
// keeps its full 32 B fragments (256 VGPRs) register-resident.
__global__ __launch_bounds__(256, 1) void k_main(const float* __restrict__ x,
                                                 const int* __restrict__ mask,
                                                 const float* __restrict__ b_a,
                                                 const uint4* __restrict__ waF,
                                                 const uint4* __restrict__ wcF,
                                                 const float* __restrict__ cvec,
                                                 unsigned short* __restrict__ feats,
                                                 float* __restrict__ partials) {
    __shared__ uint4 sA [8 * 32 * 2];   // 8KB  : a fragments   (16-bit A layout)
    __shared__ uint4 sAS[8 * 32 * 2];   // 8KB  : sim*a fragments
    __shared__ uint4 sV [16 * 32 * 2];  // 16KB : v fragments
    __shared__ float sSim[16];
    __shared__ int   sMask[16];

    const int tid  = threadIdx.x;
    const int l    = tid & 31;
    const int w    = tid >> 5;              // wave 0..7
    const int grp  = blockIdx.x >> 1;       // row group 0..1023
    const int half = blockIdx.x & 1;        // channel half
    const int nt   = half * 8 + w;          // this wave's N-tile 0..15
    const int nlo  = l & 15;
    const int ch   = nt * 16 + nlo;         // output channel
    const float ba = b_a[ch];
    const float cc = cvec[ch];

    // -------- preload this wave's 32 B fragments into registers --------
    v16bf Ba[8];    // a @ W_a          (K = 256)
    v16bf Bc[24];   // [s*a, v] @ W_cat (K = 768)
    #pragma unroll
    for (int kt = 0; kt < 8; ++kt)
        Ba[kt] = ldFrag(&waF[((size_t)(nt * 8 + kt) * 32 + l) * 2]);
    #pragma unroll
    for (int kt = 0; kt < 24; ++kt)
        Bc[kt] = ldFrag(&wcF[((size_t)(nt * 24 + kt) * 32 + l) * 2]);

    float sum = 0.f, sq = 0.f;

    for (int it = 0; it < 8; ++it) {
        const int tile0 = (grp * 8 + it) * 16;
        __syncthreads();
        // ---- phase 0: wave w computes cosine sim of rows 2w, 2w+1 ----
        {
            int r = 2 * w + (l >> 4);
            const float* base = x + (size_t)(tile0 + r) * 2048;
            const float4* qa = (const float4*)(base + 1024 + nlo * 32);
            const float4* qv = (const float4*)(base + 1536 + nlo * 32);
            float dot = 0.f, na = 0.f, nv = 0.f;
            #pragma unroll
            for (int i = 0; i < 8; ++i) {
                float4 a = qa[i], b = qv[i];
                dot += a.x*b.x + a.y*b.y + a.z*b.z + a.w*b.w;
                na  += a.x*a.x + a.y*a.y + a.z*a.z + a.w*a.w;
                nv  += b.x*b.x + b.y*b.y + b.z*b.z + b.w*b.w;
            }
            #pragma unroll
            for (int m = 1; m < 16; m <<= 1) {
                dot += __shfl_xor(dot, m, 32);
                na  += __shfl_xor(na,  m, 32);
                nv  += __shfl_xor(nv,  m, 32);
            }
            if (nlo == 0) sSim[r] = dot / fmaxf(sqrtf(na) * sqrtf(nv), COS_EPS);
            if (tid < 16) sMask[tid] = mask[tile0 + tid];
        }
        __syncthreads();
        // ---- phase 1: build A fragments (16-bit A 16x32 layout) ----
        {
            const int m  = nlo;                // row in tile
            const int kb = (l < 16) ? 0 : 8;   // K base per lane half
            const int kt = w;                  // 0..7
            float s = sSim[m];
            const float4* ra = (const float4*)(x + (size_t)(tile0 + m) * 2048 + kt * 32 + kb);
            float4 q0 = ra[0], q1 = ra[1], q2 = ra[4], q3 = ra[5];
            int idx = (kt * 32 + l) * 2;
            sA[idx]     = make_uint4(pk2(q0.x,q0.y), pk2(q0.z,q0.w), pk2(q1.x,q1.y), pk2(q1.z,q1.w));
            sA[idx + 1] = make_uint4(pk2(q2.x,q2.y), pk2(q2.z,q2.w), pk2(q3.x,q3.y), pk2(q3.z,q3.w));
            sAS[idx]     = make_uint4(pk2(s*q0.x,s*q0.y), pk2(s*q0.z,s*q0.w),
                                      pk2(s*q1.x,s*q1.y), pk2(s*q1.z,s*q1.w));
            sAS[idx + 1] = make_uint4(pk2(s*q2.x,s*q2.y), pk2(s*q2.z,s*q2.w),
                                      pk2(s*q3.x,s*q3.y), pk2(s*q3.z,s*q3.w));
            #pragma unroll
            for (int j = 0; j < 2; ++j) {
                int kv = kt + 8 * j;           // 0..15
                const float4* rv = (const float4*)(x + (size_t)(tile0 + m) * 2048 + 512 + kv * 32 + kb);
                float4 p0 = rv[0], p1 = rv[1], p2 = rv[4], p3 = rv[5];
                int iv = (kv * 32 + l) * 2;
                sV[iv]     = make_uint4(pk2(p0.x,p0.y), pk2(p0.z,p0.w), pk2(p1.x,p1.y), pk2(p1.z,p1.w));
                sV[iv + 1] = make_uint4(pk2(p2.x,p2.y), pk2(p2.z,p2.w), pk2(p3.x,p3.y), pk2(p3.z,p3.w));
            }
        }
        __syncthreads();
        // ---- phase 2: 32 WMMAs per wave, B operands in registers ----
        v8f aA = {}, aF = {};
        #pragma unroll
        for (int kt = 0; kt < 8; ++kt) {
            v16bf af = ldFrag(&sA[(kt * 32 + l) * 2]);
            aA = wmma_bf(af, Ba[kt], aA);
            v16bf sf = ldFrag(&sAS[(kt * 32 + l) * 2]);
            aF = wmma_bf(sf, Bc[kt], aF);
        }
        #pragma unroll
        for (int kv = 0; kv < 16; ++kv) {
            v16bf vf = ldFrag(&sV[(kv * 32 + l) * 2]);
            aF = wmma_bf(vf, Bc[8 + kv], aF);
        }
        // ---- bias + mask select + stats + bf16 store ----
        const int mh = (l >> 4) * 8;
        #pragma unroll
        for (int r = 0; r < 8; ++r) {
            int m   = r + mh;
            int row = tile0 + m;
            bool msk = sMask[m] != 0;
            float f = msk ? (aA[r] + ba) : (aF[r] + cc);
            sum += f; sq += f * f;
            feats[(size_t)row * 256 + ch] = f2bf(f);
        }
    }
    // combine lane halves (rows 0-7 with 8-15), write per-group partials.
    // The two half-blocks of a group write disjoint 128-channel slices.
    sum += __shfl_xor(sum, 16, 32);
    sq  += __shfl_xor(sq,  16, 32);
    if (l < 16) {
        float* p = partials + (size_t)grp * 512;
        p[ch]       = sum;     // ch already includes half*128
        p[256 + ch] = sq;
    }
}

// ---------- K2a: partial reduce (1024 -> 32 partials) ----------
__global__ __launch_bounds__(256) void k_red(const float* __restrict__ partials,
                                             float* __restrict__ p2) {
    int b = blockIdx.x;     // 0..31
    int t = threadIdx.x;    // 0..255
    float s = 0.f, q = 0.f;
    for (int i = 0; i < 32; ++i) {
        const float* p = partials + (size_t)(b * 32 + i) * 512;
        s += p[t]; q += p[256 + t];
    }
    p2[(size_t)b * 512 + t]       = s;
    p2[(size_t)b * 512 + 256 + t] = q;
}

// ---------- K2b: final reduce -> fused BN scale/shift ----------
__global__ __launch_bounds__(256) void k_bn(const float* __restrict__ p2,
                                            const float* __restrict__ gamma,
                                            const float* __restrict__ beta,
                                            float* __restrict__ bnp) {
    int c = threadIdx.x;
    float s = 0.f, q = 0.f;
    for (int b = 0; b < 32; ++b) {
        s += p2[(size_t)b * 512 + c];
        q += p2[(size_t)b * 512 + 256 + c];
    }
    float inv = 1.f / (float)N_ROWS;
    float mu  = s * inv;
    float var = fmaxf(q * inv - mu * mu, 0.f);
    float g   = gamma[c] * rsqrtf(var + BN_EPS);
    bnp[c]       = g;
    bnp[256 + c] = beta[c] - mu * g;
}

// ---------- K3: affine+ReLU + 256->2 projection (4 threads/row) ----------
__global__ __launch_bounds__(256) void k_out(const unsigned short* __restrict__ feats,
                                             const float* __restrict__ bnp,
                                             const float* __restrict__ wfc,
                                             const float* __restrict__ bfc,
                                             float* __restrict__ out) {
    __shared__ float sg[256], sb[256], sw0[256], sw1[256];
    int t = threadIdx.x;
    sg[t] = bnp[t]; sb[t] = bnp[256 + t];
    sw0[t] = wfc[2 * t]; sw1[t] = wfc[2 * t + 1];
    __syncthreads();
    int rl   = t >> 2;                       // row within block (0..63)
    int part = t & 3;                        // 64-channel segment
    size_t row = (size_t)blockIdx.x * 64 + rl;
    const uint4* fp = (const uint4*)(feats + row * 256 + part * 64);
    float a0 = 0.f, a1 = 0.f;
    #pragma unroll
    for (int i = 0; i < 8; ++i) {
        uint4 q = fp[i];
        unsigned int uu[4] = {q.x, q.y, q.z, q.w};
        #pragma unroll
        for (int j = 0; j < 4; ++j) {
            int k = part * 64 + i * 8 + j * 2;
            float h0 = fmaxf(bflo(uu[j]) * sg[k]     + sb[k],     0.f);
            float h1 = fmaxf(bfhi(uu[j]) * sg[k + 1] + sb[k + 1], 0.f);
            a0 += h0 * sw0[k] + h1 * sw0[k + 1];
            a1 += h0 * sw1[k] + h1 * sw1[k + 1];
        }
    }
    a0 += __shfl_xor(a0, 1, 32); a1 += __shfl_xor(a1, 1, 32);
    a0 += __shfl_xor(a0, 2, 32); a1 += __shfl_xor(a1, 2, 32);
    if (part == 0) {
        out[row * 2]     = a0 + bfc[0];
        out[row * 2 + 1] = a1 + bfc[1];
    }
}

extern "C" void kernel_launch(void* const* d_in, const int* in_sizes, int n_in,
                              void* d_out, int out_size, void* d_ws, size_t ws_size,
                              hipStream_t stream) {
    const float* x     = (const float*)d_in[0];
    const int*   mask  = (const int*)d_in[1];
    const float* W_a   = (const float*)d_in[2];
    const float* b_a   = (const float*)d_in[3];
    const float* W_v   = (const float*)d_in[4];
    const float* b_v   = (const float*)d_in[5];
    const float* W_f   = (const float*)d_in[6];
    const float* b_f   = (const float*)d_in[7];
    const float* gamma = (const float*)d_in[8];
    const float* beta  = (const float*)d_in[9];
    const float* W_fc  = (const float*)d_in[10];
    const float* b_fc  = (const float*)d_in[11];
    float* out = (float*)d_out;

    char* ws = (char*)d_ws;
    float*          Wvf      = (float*)(ws + 0);                // 524288
    float*          cvec     = (float*)(ws + 524288);           // 1024
    unsigned int*   waF      = (unsigned int*)(ws + 525312);    // 131072
    unsigned int*   wcF      = (unsigned int*)(ws + 656384);    // 393216
    float*          partials = (float*)(ws + 1049600);          // 1024*512*4 = 2097152
    float*          p2       = (float*)(ws + 3146752);          // 32*512*4   = 65536
    float*          bnp      = (float*)(ws + 3212288);          // 2048
    unsigned short* feats    = (unsigned short*)(ws + 3214336); // 67108864

    k_wvf <<<512,  256, 0, stream>>>(W_v, W_f, Wvf);
    k_cvec<<<1,    256, 0, stream>>>(b_v, W_f, b_f, cvec);
    k_pack<<<512,  256, 0, stream>>>(W_a, W_f, Wvf, waF, wcF);
    k_main<<<2048, 256, 0, stream>>>(x, mask, b_a,
                                     (const uint4*)waF, (const uint4*)wcF,
                                     cvec, feats, partials);
    k_red <<<32,   256, 0, stream>>>(partials, p2);
    k_bn  <<<1,    256, 0, stream>>>(p2, gamma, beta, bnp);
    k_out <<<2048, 256, 0, stream>>>(feats, bnp, W_fc, b_fc, out);
}
